// Encoder_40175124086906
// MI455X (gfx1250) — compile-verified, matrix operand order checked by reading, not run on hardware
//
#include <hip/hip_runtime.h>

// ---------------- types / helpers ----------------
typedef __attribute__((ext_vector_type(16))) __bf16 v16bf;
typedef __attribute__((ext_vector_type(8)))  float  v8f;

struct __attribute__((aligned(16))) U4 { unsigned x, y, z, w; };

union Frag {
  U4       q[2];
  unsigned u[8];
  v16bf    v;
};

__device__ __forceinline__ unsigned short f2bf(float f) {
  unsigned u = __float_as_uint(f);
  unsigned r = 0x7fffu + ((u >> 16) & 1u);
  return (unsigned short)((u + r) >> 16);
}
__device__ __forceinline__ float bf2f(unsigned short s) {
  return __uint_as_float(((unsigned)s) << 16);
}
__device__ __forceinline__ float sigf(float x)    { return 1.0f / (1.0f + __expf(-x)); }
__device__ __forceinline__ float tanhfast(float x){ return 2.0f / (1.0f + __expf(-2.0f * x)) - 1.0f; }

// ---------------- problem constants ----------------
#define EDIM  300
#define EPAD  320
#define HDIM  512
#define GDIM  2048
#define BSZ   128
#define TLEN  128
#define ROWS  16384            // BSZ*TLEN
#define OUTD  4096             // 8*H

// workspace layout (bytes)
#define OFF_WIHT ((size_t)0)
#define SZ_WIHT  ((size_t)2*EPAD*GDIM*2)
#define OFF_WHHT (OFF_WIHT + SZ_WIHT)
#define SZ_WHHT  ((size_t)2*HDIM*GDIM*2)
#define OFF_BIAS (OFF_WHHT + SZ_WHHT)
#define SZ_BIAS  ((size_t)2*GDIM*4)
#define OFF_X    (OFF_BIAS + SZ_BIAS)
#define SZ_X     ((size_t)2*ROWS*EPAD*2)
#define OFF_GX   (OFF_X + SZ_X)
#define SZ_GX    ((size_t)2*2*ROWS*GDIM*2)
#define OFF_E    (OFF_GX + SZ_GX)
#define SZ_E     ((size_t)BSZ*TLEN*TLEN*4)
#define OFF_H    (OFF_E + SZ_E)
#define SZ_H     ((size_t)4*2*BSZ*HDIM*2)
#define OFF_CNT  (OFF_H + SZ_H)

#define SCAN_WGS 32

// ---------------- kernel 1: weight prep ----------------
__global__ __launch_bounds__(256) void prep_kernel(
    const float* __restrict__ Wih_f, const float* __restrict__ Whh_f,
    const float* __restrict__ bih_f, const float* __restrict__ bhh_f,
    const float* __restrict__ Wih_b, const float* __restrict__ Whh_b,
    const float* __restrict__ bih_b, const float* __restrict__ bhh_b,
    unsigned short* __restrict__ WihT, unsigned short* __restrict__ WhhT,
    float* __restrict__ bias)
{
  const size_t N1 = (size_t)2 * EPAD * GDIM;
  const size_t N2 = (size_t)2 * HDIM * GDIM;
  const size_t N3 = (size_t)2 * GDIM;
  const size_t tot = N1 + N2 + N3;
  for (size_t i = (size_t)blockIdx.x * blockDim.x + threadIdx.x; i < tot;
       i += (size_t)gridDim.x * blockDim.x) {
    if (i < N1) {
      size_t j = i;
      int dir = (int)(j / ((size_t)EPAD * GDIM));
      j %= (size_t)EPAD * GDIM;
      int k = (int)(j / GDIM), n = (int)(j % GDIM);
      const float* W = dir ? Wih_b : Wih_f;            // [2048][300]
      WihT[i] = f2bf((k < EDIM) ? W[(size_t)n * EDIM + k] : 0.0f);
    } else if (i < N1 + N2) {
      size_t j = i - N1;
      int dir = (int)(j / ((size_t)HDIM * GDIM));
      j %= (size_t)HDIM * GDIM;
      int k = (int)(j / GDIM), n = (int)(j % GDIM);
      const float* W = dir ? Whh_b : Whh_f;            // [2048][512]
      WhhT[i - N1] = f2bf(W[(size_t)n * HDIM + k]);
    } else {
      size_t j = i - N1 - N2;
      int dir = (int)(j / GDIM);
      int n = (int)(j % GDIM);
      bias[j] = dir ? (bih_b[n] + bhh_b[n]) : (bih_f[n] + bhh_f[n]);
    }
  }
}

// ---------------- kernel 2: embedding gather -> bf16 padded X ----------------
__global__ __launch_bounds__(256) void embed_kernel(
    const int* __restrict__ A, const int* __restrict__ B,
    const float* __restrict__ embed, unsigned short* __restrict__ X)
{
  const size_t tot = (size_t)2 * ROWS * EPAD;
  for (size_t i = (size_t)blockIdx.x * blockDim.x + threadIdx.x; i < tot;
       i += (size_t)gridDim.x * blockDim.x) {
    int e = (int)(i % EPAD);
    size_t r = i / EPAD;
    int seq = (int)(r >> 14);
    int rr = (int)(r & (ROWS - 1));
    int tok = seq ? B[rr] : A[rr];
    X[i] = f2bf((e < EDIM) ? embed[(size_t)tok * EDIM + e] : 0.0f);
  }
}

// ---------------- kernel 3: input projection GEMM (bf16 WMMA) ----------------
__global__ __launch_bounds__(256) void input_proj_kernel(
    const unsigned short* __restrict__ X,    // [32768][320]
    const unsigned short* __restrict__ WihT, // [2][320][2048]
    unsigned short* __restrict__ Gx)         // [2][32768][2048]
{
  __shared__ unsigned short Xt[128 * 32];
  __shared__ unsigned short Wt[64 * 32];
  const int m0 = blockIdx.x * 128;
  const int n0 = blockIdx.y * 64;
  const int dir = blockIdx.z;
  const int tid = threadIdx.x, lane = tid & 31, wave = tid >> 5;
  const unsigned short* W = WihT + (size_t)dir * EPAD * GDIM;

  v8f acc[4];
#pragma unroll
  for (int nt = 0; nt < 4; ++nt) { v8f z = {0,0,0,0,0,0,0,0}; acc[nt] = z; }

  const int kb = (lane >= 16) ? 8 : 0;
  for (int kc = 0; kc < EPAD / 32; ++kc) {
    __syncthreads();
    for (int i = tid; i < 128 * 16; i += 256) {   // X tile [128][32], dword coalesced
      int r = i >> 4, c = i & 15;
      ((unsigned*)Xt)[r * 16 + c] =
          *(const unsigned*)(X + (size_t)(m0 + r) * EPAD + kc * 32 + c * 2);
    }
    for (int i = tid; i < 64 * 32; i += 256) {    // W tile transposed [n][k]
      int n = i & 63, k = i >> 6;
      Wt[n * 32 + k] = W[(size_t)(kc * 32 + k) * GDIM + n0 + n];
    }
    __syncthreads();

    Frag Af;
    {
      const unsigned short* xr = Xt + (wave * 16 + (lane & 15)) * 32;
      Af.q[0] = *(const U4*)(xr + kb);
      Af.q[1] = *(const U4*)(xr + kb + 16);
    }
#pragma unroll
    for (int nt = 0; nt < 4; ++nt) {
      Frag Bf;
      const unsigned short* wr = Wt + (nt * 16 + (lane & 15)) * 32;
      Bf.q[0] = *(const U4*)(wr + kb);
      Bf.q[1] = *(const U4*)(wr + kb + 16);
      acc[nt] = __builtin_amdgcn_wmma_f32_16x16x32_bf16(
          false, Af.v, false, Bf.v, (short)0, acc[nt], false, false);
    }
  }

  unsigned short* G = Gx + (size_t)dir * 2 * ROWS * GDIM;
#pragma unroll
  for (int nt = 0; nt < 4; ++nt)
#pragma unroll
    for (int r = 0; r < 8; ++r) {
      int m = m0 + wave * 16 + r + ((lane >= 16) ? 8 : 0);
      int n = n0 + nt * 16 + (lane & 15);
      G[(size_t)m * GDIM + n] = f2bf(acc[nt][r]);
    }
}

// ---------------- kernel 4: persistent cooperative BiLSTM scan ----------------
__global__ __launch_bounds__(256) void lstm_scan_kernel(
    const unsigned short* __restrict__ Gx,   // [2][32768][2048]
    const unsigned short* __restrict__ WhhT, // [2][512][2048]
    const float* __restrict__ bias,          // [2][2048]
    unsigned short* __restrict__ hbuf,       // [4][2][128][512]
    float* __restrict__ out,                 // [2][128][128][4096]
    unsigned* __restrict__ bar_cnt)
{
  extern __shared__ char smem[];
  unsigned short* frag = (unsigned short*)smem;  // [16 kc][16 gnt][32 lane][16] = 256KB

  const int wg = blockIdx.x;
  const int scan = wg >> 3, hb = wg & 7;
  const int seq = scan >> 1, dir = scan & 1;
  const int tid = threadIdx.x, lane = tid & 31, wave = tid >> 5;

  const unsigned short* WhhD = WhhT + (size_t)dir * HDIM * GDIM;
  const unsigned short* GxD  = Gx + (size_t)dir * 2 * ROWS * GDIM;
  const float* biasD = bias + dir * GDIM;
  unsigned short* hbase = hbuf + (size_t)scan * 2 * BSZ * HDIM;
  float* outBlk = out + (size_t)seq * BSZ * TLEN * OUTD;

  // one-time Whh fragment swizzle into LDS
  for (int idx = tid; idx < 16 * 16 * 32 * 16; idx += 256) {
    int e   = idx & 15;
    int ln  = (idx >> 4) & 31;
    int gnt = (idx >> 9) & 15;
    int kc  = idx >> 13;
    int gate = gnt >> 2, nt = gnt & 3;
    int K = kc * 32 + ((e >= 8) ? 16 : 0) + ((ln >= 16) ? 8 : 0) + (e & 7);
    int col = gate * 512 + hb * 64 + nt * 16 + (ln & 15);
    frag[idx] = WhhD[(size_t)K * GDIM + col];
  }

  float biasReg[16];
#pragma unroll
  for (int g = 0; g < 4; ++g)
#pragma unroll
    for (int nt = 0; nt < 4; ++nt)
      biasReg[g * 4 + nt] = biasD[g * 512 + hb * 64 + nt * 16 + (lane & 15)];

  v8f cacc[4];
#pragma unroll
  for (int nt = 0; nt < 4; ++nt) { v8f z = {0,0,0,0,0,0,0,0}; cacc[nt] = z; }

  const int rowA = wave * 16 + (lane & 15);
  const int rowC = wave * 16 + ((lane >= 16) ? 8 : 0);
  const int kbh  = (lane >= 16) ? 8 : 0;

  __syncthreads();
  unsigned phase = 0;

  for (int t = 0; t < TLEN; ++t) {
    const int tt = dir ? (TLEN - 1 - t) : t;
    const unsigned short* hcur = hbase + (size_t)(t & 1) * BSZ * HDIM;
    unsigned short* hnxt = hbase + (size_t)((t + 1) & 1) * BSZ * HDIM;

    // acc = bias + Gx[:, tt, gate-cols]
    v8f acc[4][4];
#pragma unroll
    for (int g = 0; g < 4; ++g)
#pragma unroll
      for (int nt = 0; nt < 4; ++nt) {
        const int col = g * 512 + hb * 64 + nt * 16 + (lane & 15);
        v8f a;
#pragma unroll
        for (int r = 0; r < 8; ++r) {
          int b = rowC + r;
          size_t grow = (size_t)seq * ROWS + (size_t)b * TLEN + tt;
          a[r] = biasReg[g * 4 + nt] + bf2f(GxD[grow * GDIM + col]);
        }
        acc[g][nt] = a;
      }

    if (t + 1 < TLEN) {  // prefetch next timestep's gate slab
      int tn = dir ? (TLEN - 2 - t) : (t + 1);
      __builtin_prefetch(
          &GxD[((size_t)seq * ROWS + (size_t)rowA * TLEN + tn) * GDIM + hb * 64 + (lane & 15)],
          0, 1);
    }

    // gates += h_prev @ Whh^T  (K = 512)
    for (int kc = 0; kc < 16; ++kc) {
      Frag Af;
      const unsigned short* hrow = hcur + (size_t)rowA * HDIM;
      const int kbase = kc * 32 + kbh;
#pragma unroll
      for (int p = 0; p < 8; ++p) {
        int k = kbase + ((p >= 4) ? 16 : 0) + (p & 3) * 2;
        Af.u[p] = *(const unsigned*)(hrow + k);
      }
#pragma unroll
      for (int g = 0; g < 4; ++g)
#pragma unroll
        for (int nt = 0; nt < 4; ++nt) {
          Frag Bf;
          const unsigned short* bp =
              frag + (((size_t)kc * 16 + (g * 4 + nt)) * 32 + lane) * 16;
          Bf.q[0] = *(const U4*)(bp);
          Bf.q[1] = *(const U4*)(bp + 8);
          acc[g][nt] = __builtin_amdgcn_wmma_f32_16x16x32_bf16(
              false, Af.v, false, Bf.v, (short)0, acc[g][nt], false, false);
        }
    }

    // LSTM cell update; write h (bf16 state + fp32 bar slice of d_out)
#pragma unroll
    for (int nt = 0; nt < 4; ++nt) {
      v8f c = cacc[nt];
#pragma unroll
      for (int r = 0; r < 8; ++r) {
        float ig = sigf(acc[0][nt][r]);
        float fg = sigf(acc[1][nt][r]);
        float gg = tanhfast(acc[2][nt][r]);
        float og = sigf(acc[3][nt][r]);
        float cn = fg * c[r] + ig * gg;
        float hn = og * tanhfast(cn);
        c[r] = cn;
        int b = rowC + r;
        int col = hb * 64 + nt * 16 + (lane & 15);
        hnxt[(size_t)b * HDIM + col] = f2bf(hn);
        outBlk[((size_t)b * TLEN + tt) * OUTD + dir * HDIM + col] = hn;
      }
      cacc[nt] = c;
    }

    // device-wide barrier (monotonic counter)
    __syncthreads();
    if (tid == 0) {
      __threadfence();
      atomicAdd(bar_cnt, 1u);
      const unsigned target = (phase + 1) * SCAN_WGS;
      while (__atomic_load_n(bar_cnt, __ATOMIC_RELAXED) < target)
        __builtin_amdgcn_s_sleep(2);
    }
    ++phase;
    __syncthreads();
    __threadfence();
  }
}

// ---------------- kernel 5: attention scores E = A_bar . B_bar^T ----------------
__global__ __launch_bounds__(256) void attn_scores_kernel(
    const float* __restrict__ out, float* __restrict__ E)
{
  __shared__ unsigned short At[128 * 32];
  __shared__ unsigned short Bt[128 * 32];
  const int b = blockIdx.x;
  const int tid = threadIdx.x, lane = tid & 31, wave = tid >> 5;
  const float* Abar = out + (size_t)b * TLEN * OUTD;
  const float* Bbar = out + (size_t)BSZ * TLEN * OUTD + (size_t)b * TLEN * OUTD;

  v8f acc[8];
#pragma unroll
  for (int nt = 0; nt < 8; ++nt) { v8f z = {0,0,0,0,0,0,0,0}; acc[nt] = z; }

  const int kb = (lane >= 16) ? 8 : 0;
  for (int kc = 0; kc < 32; ++kc) {        // K = 1024 = 2H
    __syncthreads();
    for (int i = tid; i < 4096; i += 256) {
      int r = i >> 5, k = i & 31;
      At[r * 32 + k] = f2bf(Abar[(size_t)r * OUTD + kc * 32 + k]);
      Bt[r * 32 + k] = f2bf(Bbar[(size_t)r * OUTD + kc * 32 + k]);
    }
    __syncthreads();

    Frag Af;
    {
      const unsigned short* ar = At + (wave * 16 + (lane & 15)) * 32;
      Af.q[0] = *(const U4*)(ar + kb);
      Af.q[1] = *(const U4*)(ar + kb + 16);
    }
#pragma unroll
    for (int nt = 0; nt < 8; ++nt) {
      Frag Bf;
      const unsigned short* br = Bt + (nt * 16 + (lane & 15)) * 32;
      Bf.q[0] = *(const U4*)(br + kb);
      Bf.q[1] = *(const U4*)(br + kb + 16);
      acc[nt] = __builtin_amdgcn_wmma_f32_16x16x32_bf16(
          false, Af.v, false, Bf.v, (short)0, acc[nt], false, false);
    }
  }

#pragma unroll
  for (int nt = 0; nt < 8; ++nt)
#pragma unroll
    for (int r = 0; r < 8; ++r) {
      int i = wave * 16 + r + ((lane >= 16) ? 8 : 0);
      int j = nt * 16 + (lane & 15);
      E[(size_t)b * TLEN * TLEN + (size_t)i * TLEN + j] = acc[nt][r];
    }
}

// ---------------- kernel 6: fused softmax + tilde GEMM + enhancement ----------------
// mode 0: A_tilde = rowsoftmax(E) @ B_bar ; mode 1: B_tilde = colsoftmax(E)^T @ A_bar
__global__ __launch_bounds__(256) void attn_tilde_kernel(
    float* __restrict__ out, const float* __restrict__ E)
{
  extern __shared__ char smem[];
  unsigned short* P  = (unsigned short*)smem;             // [128][128] bf16 exp()
  unsigned short* Bt = (unsigned short*)(smem + 32768);   // [256][128] bf16 (transposed)
  float* sums = (float*)(smem + 32768 + 65536);           // [128]

  const int b = blockIdx.x, nchunk = blockIdx.y, mode = blockIdx.z;
  const int d0 = nchunk * 256;
  const int tid = threadIdx.x, lane = tid & 31, wave = tid >> 5;
  const float* Eb = E + (size_t)b * TLEN * TLEN;
  float* blkA = out + (size_t)b * TLEN * OUTD;
  float* blkB = out + (size_t)BSZ * TLEN * OUTD + (size_t)b * TLEN * OUTD;
  float* dstBlk = (mode == 0) ? blkA : blkB;        // enhancement target + bar source
  const float* srcB = (mode == 0) ? blkB : blkA;    // matrix summed over

  if (tid < 128) {  // unnormalized softmax row into P, sum into sums[]
    float mx = -3.0e38f;
    for (int k = 0; k < 128; ++k) {
      float v = (mode == 0) ? Eb[tid * 128 + k] : Eb[k * 128 + tid];
      mx = fmaxf(mx, v);
    }
    float s = 0.0f;
    for (int k = 0; k < 128; ++k) {
      float v = (mode == 0) ? Eb[tid * 128 + k] : Eb[k * 128 + tid];
      float e = __expf(v - mx);
      s += e;
      P[tid * 128 + k] = f2bf(e);
    }
    sums[tid] = s;
  }
  for (int i = tid; i < 128 * 256; i += 256) {  // Bt[d][k] = srcB[k][d0+d]
    int k = i >> 8, d = i & 255;
    Bt[d * 128 + k] = f2bf(srcB[(size_t)k * OUTD + d0 + d]);
  }
  __syncthreads();

  v8f acc[16];
#pragma unroll
  for (int nt = 0; nt < 16; ++nt) { v8f z = {0,0,0,0,0,0,0,0}; acc[nt] = z; }

  const int kb = (lane >= 16) ? 8 : 0;
  for (int kc = 0; kc < 4; ++kc) {              // K = 128
    Frag Af;
    {
      const unsigned short* pr = P + (wave * 16 + (lane & 15)) * 128 + kc * 32;
      Af.q[0] = *(const U4*)(pr + kb);
      Af.q[1] = *(const U4*)(pr + kb + 16);
    }
#pragma unroll
    for (int nt = 0; nt < 16; ++nt) {
      Frag Bf;
      const unsigned short* br = Bt + (nt * 16 + (lane & 15)) * 128 + kc * 32;
      Bf.q[0] = *(const U4*)(br + kb);
      Bf.q[1] = *(const U4*)(br + kb + 16);
      acc[nt] = __builtin_amdgcn_wmma_f32_16x16x32_bf16(
          false, Af.v, false, Bf.v, (short)0, acc[nt], false, false);
    }
  }

#pragma unroll
  for (int nt = 0; nt < 16; ++nt)
#pragma unroll
    for (int r = 0; r < 8; ++r) {
      int row = wave * 16 + r + ((lane >= 16) ? 8 : 0);
      int d = d0 + nt * 16 + (lane & 15);
      float tl = acc[nt][r] * (1.0f / sums[row]);
      float bar = dstBlk[(size_t)row * OUTD + d];
      dstBlk[(size_t)row * OUTD + 1024 + d] = tl;
      dstBlk[(size_t)row * OUTD + 2048 + d] = bar - tl;
      dstBlk[(size_t)row * OUTD + 3072 + d] = bar * tl;
    }
}

// ---------------- host launcher ----------------
extern "C" void kernel_launch(void* const* d_in, const int* in_sizes, int n_in,
                              void* d_out, int out_size, void* d_ws, size_t ws_size,
                              hipStream_t stream) {
  const int*   A     = (const int*)d_in[0];
  const int*   Bi    = (const int*)d_in[1];
  const float* embed = (const float*)d_in[2];
  const float* Wih_f = (const float*)d_in[3];
  const float* Whh_f = (const float*)d_in[4];
  const float* bih_f = (const float*)d_in[5];
  const float* bhh_f = (const float*)d_in[6];
  const float* Wih_b = (const float*)d_in[7];
  const float* Whh_b = (const float*)d_in[8];
  const float* bih_b = (const float*)d_in[9];
  const float* bhh_b = (const float*)d_in[10];

  char* ws = (char*)d_ws;
  unsigned short* WihT = (unsigned short*)(ws + OFF_WIHT);
  unsigned short* WhhT = (unsigned short*)(ws + OFF_WHHT);
  float*          bias = (float*)(ws + OFF_BIAS);
  unsigned short* X    = (unsigned short*)(ws + OFF_X);
  unsigned short* Gx   = (unsigned short*)(ws + OFF_GX);
  float*          Em   = (float*)(ws + OFF_E);
  unsigned short* hbuf = (unsigned short*)(ws + OFF_H);
  unsigned*       cnt  = (unsigned*)(ws + OFF_CNT);
  float*          outp = (float*)d_out;

  (void)hipFuncSetAttribute((const void*)lstm_scan_kernel,
                            hipFuncAttributeMaxDynamicSharedMemorySize, 262144);
  (void)hipFuncSetAttribute((const void*)attn_tilde_kernel,
                            hipFuncAttributeMaxDynamicSharedMemorySize, 98816 + 512);

  // zero h ping-pong buffers + barrier counter (deterministic per call)
  (void)hipMemsetAsync(ws + OFF_H, 0, SZ_H + 256, stream);

  prep_kernel<<<1024, 256, 0, stream>>>(Wih_f, Whh_f, bih_f, bhh_f,
                                        Wih_b, Whh_b, bih_b, bhh_b,
                                        WihT, WhhT, bias);
  embed_kernel<<<4096, 256, 0, stream>>>(A, Bi, embed, X);
  input_proj_kernel<<<dim3(256, 32, 2), 256, 0, stream>>>(X, WihT, Gx);
  lstm_scan_kernel<<<SCAN_WGS, 256, 262144, stream>>>(Gx, WhhT, bias, hbuf, outp, cnt);
  attn_scores_kernel<<<BSZ, 256, 0, stream>>>((const float*)d_out, Em);
  attn_tilde_kernel<<<dim3(BSZ, 4, 2), 256, 98816 + 512, stream>>>(outp, Em);
}
